// IAF_61280593380090
// MI455X (gfx1250) — matrix operand order repaired and checked by reading, not compiled, not run on hardware
//
#include <hip/hip_runtime.h>

typedef __attribute__((ext_vector_type(16))) _Float16 v16h;
typedef __attribute__((ext_vector_type(8)))  _Float16 v8h;
typedef __attribute__((ext_vector_type(8)))  float    v8f;

// ---------------------------------------------------------------------------
// Prep kernel: apply MADE masks, transpose to [N][K], convert to f16.
// Weight layout in d_ws:  half  wh[(s*4+layer)*64*64]  with wh[..][n][k]
// Bias  layout in d_ws+64KB: float bh[(s*4+layer)*64]
// Masks (D=H=64, d_h[j] = j%63 + 1):
//   layer0 (W0*m_in)  : keep if  k      <= n%63
//   layer1/2 (W*m_hid): keep if  k%63   <= n%63
//   layer3 (Wout*m_out): keep if k%63   <  n
// ---------------------------------------------------------------------------
__global__ void iaf_prep(const float* __restrict__ W0, const float* __restrict__ W1,
                         const float* __restrict__ W2, const float* __restrict__ Wout,
                         const float* __restrict__ b0, const float* __restrict__ b1,
                         const float* __restrict__ b2, const float* __restrict__ bout,
                         _Float16* __restrict__ wh, float* __restrict__ bh)
{
    int idx = blockIdx.x * blockDim.x + threadIdx.x;
    if (idx < 32768) {
        int k = idx & 63, n = (idx >> 6) & 63, layer = (idx >> 12) & 3, s = idx >> 14;
        float w; bool m;
        if (layer == 0)      { w = W0  [s*4096 + k*64 + n]; m = (k        <= (n % 63)); }
        else if (layer == 1) { w = W1  [s*4096 + k*64 + n]; m = ((k % 63) <= (n % 63)); }
        else if (layer == 2) { w = W2  [s*4096 + k*64 + n]; m = ((k % 63) <= (n % 63)); }
        else                 { w = Wout[s*4096 + k*64 + n]; m = ((k % 63) <  n);        }
        wh[((s*4 + layer)*64 + n)*64 + k] = (_Float16)(m ? w : 0.0f);
    } else if (idx < 32768 + 512) {
        int b = idx - 32768; int n = b & 63; int layer = (b >> 6) & 3; int s = b >> 8;
        const float* src = (layer == 0) ? b0 : (layer == 1) ? b1 : (layer == 2) ? b2 : bout;
        bh[b] = src[s*64 + n];
    }
}

// ---------------------------------------------------------------------------
// One made_shift:  mOut = mBase + sign * MADE(mIn)   (all masters f32 in LDS)
// Per wave: 4 layers x 4 N-tiles x 2 K-halves = 32 v_wmma_f32_16x16x32_f16.
// ---------------------------------------------------------------------------
__device__ __forceinline__ void made_shift(int s,
                                           const float* __restrict__ mIn,
                                           float* __restrict__ mOut,
                                           const float* __restrict__ mBase,
                                           float sign,
                                           const _Float16* __restrict__ wts,   // current stage, [layer][n][k]
                                           const float* __restrict__ bias,    // all stages
                                           _Float16* __restrict__ actAll)
{
    const int tid  = threadIdx.x;
    const int wave = tid >> 5;
    const int lane = tid & 31;
    const int hh   = lane >> 4;   // half-wave select
    const int lm   = lane & 15;
    _Float16* act = actAll + wave * (16*64);

    // f32 master strip -> f16 act strip (wave-private rows)
    for (int j = lane; j < 16*64; j += 32)
        act[j] = (_Float16)mIn[wave*16*64 + j];
    __syncthreads();

    #pragma unroll
    for (int layer = 0; layer < 4; ++layer) {
        const _Float16* wb = wts + layer * 4096;
        const float*    bb = bias + (s*4 + layer) * 64;

        // A fragments: 16x32 f16 per K-half, ISA layout.
        v16h a[2];
        #pragma unroll
        for (int kk = 0; kk < 2; ++kk) {
            const _Float16* pa = act + lm*64 + kk*32 + hh*8;
            v8h lo = *(const v8h*)pa;          // K = kbase + hh*8 .. +7
            v8h hi = *(const v8h*)(pa + 16);   // K = kbase + 16 + hh*8 .. +7
            #pragma unroll
            for (int e = 0; e < 8; ++e) { a[kk][e] = lo[e]; a[kk][e + 8] = hi[e]; }
        }

        #pragma unroll
        for (int nt = 0; nt < 4; ++nt) {
            v8f c = {};
            #pragma unroll
            for (int kk = 0; kk < 2; ++kk) {
                // B fragment: lane holds N = nt*16+lm, K = kbase + hh*8 .. +7 and kbase + 16 + hh*8 .. +7 (weights stored [N][K])
                const _Float16* pb = wb + (nt*16 + lm)*64 + kk*32 + hh*8;
                v8h blo = *(const v8h*)pb;
                v8h bhi = *(const v8h*)(pb + 16);
                v16h b;
                #pragma unroll
                for (int e = 0; e < 8; ++e) { b[e] = blo[e]; b[e + 8] = bhi[e]; }
                c = __builtin_amdgcn_wmma_f32_16x16x32_f16(false, a[kk], false, b,
                                                           (short)0, c, false, false);
            }
            float bv = bb[nt*16 + lm];
            if (layer < 3) {
                #pragma unroll
                for (int r = 0; r < 8; ++r) {          // relu, back to f16 strip
                    float v = c[r] + bv; v = v > 0.0f ? v : 0.0f;
                    int m = r + hh*8;
                    act[m*64 + nt*16 + lm] = (_Float16)v;
                }
            } else {
                #pragma unroll
                for (int r = 0; r < 8; ++r) {          // apply to f32 master
                    int m = r + hh*8;
                    int gidx = (wave*16 + m)*64 + nt*16 + lm;
                    mOut[gidx] = mBase[gidx] + sign * (c[r] + bv);
                }
            }
        }
        __syncthreads();
    }
}

// ---------------------------------------------------------------------------
// Main kernel: 64 threads (2 wave32) per block, 32 batch rows per block.
// ---------------------------------------------------------------------------
__global__ __launch_bounds__(64) void iaf_main(const float* __restrict__ zm,
                                               const float* __restrict__ lv,
                                               const float* __restrict__ eps,
                                               const _Float16* __restrict__ gw,
                                               const float* __restrict__ gb,
                                               float* __restrict__ out)
{
    __shared__ __align__(16) _Float16 s_wts[4*64*64];  // 32 KB, current stage
    __shared__ __align__(16) float    s_bias[512];     //  2 KB, both stages
    __shared__ __align__(16) _Float16 s_act[2*16*64];  //  4 KB, f16 strips
    __shared__ __align__(16) float    s_m1[32*64];     //  8 KB, z / x master
    __shared__ __align__(16) float    s_m2[32*64];     //  8 KB, u master

    const int tid = threadIdx.x;
    const int r0  = blockIdx.x * 32;

    for (int j = tid; j < 512; j += 64) s_bias[j] = gb[j];

    // z = z_mean + exp(0.5*logvar)*eps
    for (int j = tid; j < 2048; j += 64) {
        int g = r0*64 + j;
        s_m1[j] = zm[g] + __expf(0.5f * lv[g]) * eps[g];
    }
    // stage-1 weights
    {
        const uint4* src = (const uint4*)(gw + 16384);
        uint4* dst = (uint4*)s_wts;
        for (int j = tid; j < 2048; j += 64) dst[j] = src[j];
    }
    __syncthreads();

    // ---- inverse pass:  z -= shift(z, s=1); reverse; z -= shift(z, s=0) ----
    made_shift(1, s_m1, s_m1, s_m1, -1.0f, s_wts, s_bias, s_act);
    {
        float t[32];
        #pragma unroll
        for (int i = 0; i < 32; ++i) {
            int j = tid + i*64; int m = j >> 6, k = j & 63;
            t[i] = s_m1[m*64 + (63 - k)];
        }
        __syncthreads();
        #pragma unroll
        for (int i = 0; i < 32; ++i) s_m1[tid + i*64] = t[i];
    }
    {
        const uint4* src = (const uint4*)gw;           // stage-0 weights
        uint4* dst = (uint4*)s_wts;
        for (int j = tid; j < 2048; j += 64) dst[j] = src[j];
    }
    __syncthreads();
    made_shift(0, s_m1, s_m1, s_m1, -1.0f, s_wts, s_bias, s_act);

    // samples -> output, u = samples, x = 0
    for (int j = tid; j < 2048; j += 64) {
        float v = s_m1[j];
        out[r0*64 + j] = v;
        s_m2[j] = v;
        s_m1[j] = 0.0f;
    }
    __syncthreads();

    // ---- forward pass, stage 0: 64 fixed-point iterations x = u + shift(x) ----
    #pragma unroll 1
    for (int it = 0; it < 64; ++it)
        made_shift(0, s_m1, s_m1, s_m2, 1.0f, s_wts, s_bias, s_act);

    // u = reverse(x)
    for (int j = tid; j < 2048; j += 64) {
        int m = j >> 6, k = j & 63;
        s_m2[j] = s_m1[m*64 + (63 - k)];
    }
    __syncthreads();
    {
        const uint4* src = (const uint4*)(gw + 16384); // stage-1 weights
        uint4* dst = (uint4*)s_wts;
        for (int j = tid; j < 2048; j += 64) dst[j] = src[j];
    }
    for (int j = tid; j < 2048; j += 64) s_m1[j] = 0.0f;
    __syncthreads();

    // ---- forward pass, stage 1 ----
    #pragma unroll 1
    for (int it = 0; it < 64; ++it)
        made_shift(1, s_m1, s_m1, s_m2, 1.0f, s_wts, s_bias, s_act);
    // u = final x (in s_m1), no reversal on last stage

    // ---- density ----
    float* sc = (float*)s_act;   // reuse as f32 scratch
    int row = tid & 31, kh = tid >> 5;
    float slv = 0.0f, ssq = 0.0f;
    #pragma unroll 1
    for (int k2 = 0; k2 < 32; ++k2) {
        int k = kh*32 + k2;
        int g = (r0 + row)*64 + k;
        float l = lv[g];
        slv += l;
        float d = s_m1[row*64 + k] - zm[g];
        ssq += d * d * __expf(-l);
    }
    sc[tid] = slv; sc[64 + tid] = ssq;
    __syncthreads();
    if (tid < 32) {
        float sl = sc[tid] + sc[tid + 32];
        float sq = sc[64 + tid] + sc[64 + tid + 32];
        out[4096*64 + r0 + tid] = -0.5f * 64.0f * 1.8378770664093453f
                                  - 0.5f * sl - 0.5f * sq;
    }
}

// ---------------------------------------------------------------------------
extern "C" void kernel_launch(void* const* d_in, const int* in_sizes, int n_in,
                              void* d_out, int out_size, void* d_ws, size_t ws_size,
                              hipStream_t stream)
{
    const float* zm   = (const float*)d_in[0];
    const float* lv   = (const float*)d_in[1];
    const float* eps  = (const float*)d_in[2];
    const float* W0   = (const float*)d_in[3];
    const float* b0   = (const float*)d_in[4];
    const float* W1   = (const float*)d_in[5];
    const float* b1   = (const float*)d_in[6];
    const float* W2   = (const float*)d_in[7];
    const float* b2   = (const float*)d_in[8];
    const float* Wout = (const float*)d_in[9];
    const float* bout = (const float*)d_in[10];

    _Float16* wh = (_Float16*)d_ws;                    // 64 KB masked f16 weights
    float*    bh = (float*)((char*)d_ws + 65536);      //  2 KB biases
    float*    out = (float*)d_out;

    iaf_prep<<<(33280 + 127) / 128, 128, 0, stream>>>(W0, W1, W2, Wout,
                                                      b0, b1, b2, bout, wh, bh);
    iaf_main<<<128, 64, 0, stream>>>(zm, lv, eps, wh, bh, out);
}
